// Hierarchical_Brand_Attention_65601330479326
// MI455X (gfx1250) — matrix operand order repaired and checked by reading, not compiled, not run on hardware
//
#include <hip/hip_runtime.h>

#define BB 4096
#define SS 200
#define DD 256
#define STILES 13            // ceil(200/16)
#define BPB 8                // batch rows per block
#define NW1 65536            // 256*256
#define TOTDW (BB * SS * DD) // total seq dwords (209,715,200 < 2^32)

typedef __attribute__((ext_vector_type(16))) __bf16    v16bf;
typedef __attribute__((ext_vector_type(8)))  float     v8f;
typedef __attribute__((ext_vector_type(4)))  unsigned  v4u;
typedef __attribute__((ext_vector_type(8)))  int       v8i;
typedef __attribute__((ext_vector_type(4)))  int       v4i;

#if __has_builtin(__builtin_amdgcn_tensor_load_to_lds) && __has_builtin(__builtin_amdgcn_s_wait_tensorcnt)
#define HAVE_TDM 1
#else
#define HAVE_TDM 0
#endif

// ---------------------------------------------------------------------------
// Prep 1: pack W1 into per-lane WMMA B-fragment layout (bf16 hi/lo planes).
// B[k][n] = W1[n][k] (torch Linear: y = x W^T). Fragment f = kt*16+nt, lane L
// owns 16 contiguous bf16: j -> K_local = (j<8) ? 8*(L>>4)+j : 8+8*(L>>4)+j, N = nt*16+(L&15).
// ---------------------------------------------------------------------------
__global__ void pack_w1_kernel(const float* __restrict__ W1,
                               __bf16* __restrict__ hi, __bf16* __restrict__ lo) {
    int tid  = blockIdx.x * 256 + threadIdx.x;     // 65536 total
    int j    = tid & 15;
    int L    = (tid >> 4) & 31;
    int frag = tid >> 9;
    int nt   = frag & 15;
    int kt   = frag >> 4;
    int kl   = (L < 16) ? ((j < 8) ? j : (8 + j)) : ((j < 8) ? (8 + j) : (16 + j));
    int d    = kt * 32 + kl;
    int e    = nt * 16 + (L & 15);
    float  x = W1[e * DD + d];
    __bf16 h = (__bf16)x;
    hi[tid] = h;
    lo[tid] = (__bf16)(x - (float)h);
}

// ---------------------------------------------------------------------------
// Prep 2: transpose W2 for coalesced reads.
// ---------------------------------------------------------------------------
__global__ void transpose_w2_kernel(const float* __restrict__ W2,
                                    float* __restrict__ W2T) {
    W2T[blockIdx.x * DD + threadIdx.x] = W2[threadIdx.x * DD + blockIdx.x];
}

// ---------------------------------------------------------------------------
// Prep 3: cate_p[b][e] = cate[b] @ W2^T + b2
// ---------------------------------------------------------------------------
struct CpShared {
    float w2t[NW1];
    float crow[DD];
};
__global__ __launch_bounds__(256, 1)
void catep_kernel(const float* __restrict__ cate, const float* __restrict__ b2,
                  const float* __restrict__ W2T, float* __restrict__ catep) {
    __shared__ CpShared sm;
    const int tid = threadIdx.x;
    {
        const float4* src = (const float4*)W2T;
        float4* dst = (float4*)sm.w2t;
        for (int i = tid; i < NW1 / 4; i += 256) dst[i] = src[i];
    }
    __syncthreads();
    for (int bi = 0; bi < 16; ++bi) {
        int b = blockIdx.x * 16 + bi;
        sm.crow[tid] = cate[(long)b * DD + tid];
        __syncthreads();
        float acc = b2[tid];
        #pragma unroll 8
        for (int d = 0; d < DD; ++d)
            acc = fmaf(sm.crow[d], sm.w2t[d * DD + tid], acc);
        catep[(long)b * DD + tid] = acc;
        __syncthreads();
    }
}

// ---------------------------------------------------------------------------
// Main fused kernel. LDS (uint4-typed for guaranteed b128 DS traffic):
//   w1hi4/w1lo4 : packed W1 B-fragments (256 KB, loaded once via TDM)
//   fstage      : f32 seq tile landing zone for async TDM prefetch (16 KB)
//   sqhi4/sqlo4 : bf16 hi/lo seq tile (A-fragments + pooling source)
// ---------------------------------------------------------------------------
struct MainShared {
    uint4  w1hi4[NW1 / 8];   // 128 KB  (struct offset 0)
    uint4  w1lo4[NW1 / 8];   // 128 KB
    float  fstage[16 * DD];  //  16 KB
    uint4  sqhi4[16 * DD / 8];
    uint4  sqlo4[16 * DD / 8];
    float  bias[DD];
    float  scores[16];
    float  wbuf[16];
};

union FragU { uint4 u[2]; v16bf v; };
union PK8   { __bf16 h[8]; uint4 u; };

#if HAVE_TDM
// Issue TENSOR_LOAD_TO_LDS for a (tile_d0 x tile_d1) dword tile.
// D# packing per CDNA5 ISA:
//   g0: [1:0]count=1 | [63:32]lds_addr | [120:64]global_addr | [127:126]type=2
//   g1: [17:16]data_size=2(4B) | [79:48]tensor_dim0 | [111:80]tensor_dim1
//       [127:112]tile_dim0 | [143:128]tile_dim1 | [207:160]tensor_dim0_stride
__device__ __forceinline__ void tdm_load(unsigned lds_off, const void* gptr,
                                         unsigned tensor_d0, unsigned tensor_d1,
                                         unsigned tile_d0, unsigned tile_d1,
                                         unsigned stride_dw) {
    unsigned long long ga = (unsigned long long)(size_t)gptr;
    v4u g0;
    g0[0] = 1u;                                   // count=1 (valid descriptor)
    g0[1] = lds_off;                              // LDS byte address
    g0[2] = (unsigned)(ga & 0xffffffffull);       // global_addr[31:0]
    g0[3] = (unsigned)((ga >> 32) & 0x01ffffffull) | (2u << 30);  // addr[56:32] | type=2
    v8i g1;
    g1[0] = (int)(2u << 16);                                        // data_size = 4B
    g1[1] = (int)((tensor_d0 & 0xffffu) << 16);                     // tensor_dim0 lo16
    g1[2] = (int)((tensor_d0 >> 16) | ((tensor_d1 & 0xffffu) << 16));
    g1[3] = (int)((tensor_d1 >> 16) | ((tile_d0 & 0xffffu) << 16)); // tile_dim0
    g1[4] = (int)(tile_d1 & 0xffffu);                               // tile_dim1 (tile_dim2=0)
    g1[5] = (int)stride_dw;                                         // dim0 stride lo32
    g1[6] = 0;
    g1[7] = 0;
    #pragma unroll
    for (int i = 0; i < 4; ++i) g0[i] = (unsigned)__builtin_amdgcn_readfirstlane((int)g0[i]);
    #pragma unroll
    for (int i = 0; i < 8; ++i) g1[i] = __builtin_amdgcn_readfirstlane(g1[i]);
    v4i gz = {0, 0, 0, 0};
#if defined(__clang_major__) && __clang_major__ >= 23
    v8i gz8 = {0, 0, 0, 0, 0, 0, 0, 0};
    __builtin_amdgcn_tensor_load_to_lds(g0, g1, gz, gz, gz8, 0);
#else
    __builtin_amdgcn_tensor_load_to_lds(g0, g1, gz, gz, 0);
#endif
}
#endif

__global__ __launch_bounds__(256, 1)
void fused_attn_kernel(const float* __restrict__ seq,
                       const unsigned char* __restrict__ mask,
                       const float* __restrict__ b1,
                       const float* __restrict__ catep,
                       const __bf16* __restrict__ w1hi_g,   // contiguous with w1lo_g
                       const float* __restrict__ w1lo_g_unused,
                       float* __restrict__ out) {
    __shared__ MainShared sm;
    const int tid  = threadIdx.x;
    const int lane = tid & 31;
    const int wave = tid >> 5;

    // ---- prologue: async W1-image copy (256 KB) + first seq tile ----
#if HAVE_TDM
    if (wave == 0) {
        // 65536 dwords as 8x8192 2-D tile into LDS offset 0 (w1hi4|w1lo4)
        tdm_load(0u, w1hi_g, 8192u, 8u, 8192u, 8u, 8192u);
        unsigned start0 = (unsigned)(blockIdx.x * BPB) * (SS * DD);
        tdm_load((unsigned)__builtin_offsetof(MainShared, fstage),
                 seq + start0, TOTDW - start0, 1u, 4096u, 1u, 4096u);
    }
#else
    {
        const uint4* sh = (const uint4*)w1hi_g;
        uint4* dh = (uint4*)sm.w1hi4;
        for (int i = tid; i < NW1 / 4; i += 256) dh[i] = sh[i];   // hi+lo contiguous
    }
    __syncthreads();
#endif

    float m = -3.0e38f, l = 0.0f, acc = 0.0f;   // online-softmax state; tid owns column d

    for (int ti = 0; ti < BPB * STILES; ++ti) {
        const int b  = blockIdx.x * BPB + ti / STILES;
        const int st = ti % STILES;
        const int s0 = st * 16;

        if (st == 0) {
            m = -3.0e38f; l = 0.0f; acc = 0.0f;
            sm.bias[tid] = b1[tid] + catep[(long)b * DD + tid];
        }

#if HAVE_TDM
        if (wave == 0) __builtin_amdgcn_s_wait_tensorcnt(0);  // tile ti (+W1 first time) landed
        __syncthreads();
#endif
        // ---- convert staged f32 tile -> bf16 hi/lo planes ----
        {
            const int r  = tid >> 4;
            const int c0 = (tid & 15) << 4;
            PK8 ph[2], pl[2];
            #pragma unroll
            for (int j4 = 0; j4 < 4; ++j4) {
#if HAVE_TDM
                float4 v = ((const float4*)sm.fstage)[(r * DD + c0) / 4 + j4];
#else
                unsigned grow = (unsigned)(b * SS + s0 + r);
                float4 v = {0.f, 0.f, 0.f, 0.f};
                if (grow < (unsigned)(BB * SS))
                    v = *(const float4*)(seq + (long)grow * DD + c0 + j4 * 4);
#endif
                float xs[4] = {v.x, v.y, v.z, v.w};
                #pragma unroll
                for (int k = 0; k < 4; ++k) {
                    int idx = j4 * 4 + k;
                    __bf16 h = (__bf16)xs[k];
                    ph[idx >> 3].h[idx & 7] = h;
                    pl[idx >> 3].h[idx & 7] = (__bf16)(xs[k] - (float)h);
                }
            }
            int u4 = r * 32 + (tid & 15) * 2;
            sm.sqhi4[u4] = ph[0].u;  sm.sqhi4[u4 + 1] = ph[1].u;
            sm.sqlo4[u4] = pl[0].u;  sm.sqlo4[u4 + 1] = pl[1].u;
        }
        if (tid < 16) sm.scores[tid] = 0.0f;
        __syncthreads();

#if HAVE_TDM
        // ---- async prefetch of next tile into fstage (now free) ----
        if (wave == 0 && ti + 1 < BPB * STILES) {
            unsigned startn = (unsigned)(blockIdx.x * BPB + (ti + 1) / STILES) * (SS * DD)
                            + (unsigned)((ti + 1) % STILES) * 16u * DD;
            tdm_load((unsigned)__builtin_offsetof(MainShared, fstage),
                     seq + startn, TOTDW - startn, 1u, 4096u, 1u, 4096u);
        }
#endif

        // ---- GEMM: wave handles n-tiles 2*wave, 2*wave+1; bf16 hi/lo x3 ----
        v8f cacc[2] = {};
        const int row   = lane & 15;
        const int khalf = (lane >> 4);             // 0 or 1 (unit: uint4 within k-tile)
        #pragma unroll
        for (int kt = 0; kt < 8; ++kt) {
            const int ai = row * 32 + kt * 4 + khalf;
            FragU ah, al;
            ah.u[0] = sm.sqhi4[ai];  ah.u[1] = sm.sqhi4[ai + 2];
            al.u[0] = sm.sqlo4[ai];  al.u[1] = sm.sqlo4[ai + 2];
            #pragma unroll
            for (int t = 0; t < 2; ++t) {
                const int nt  = wave * 2 + t;
                const int fb4 = ((kt * 16 + nt) * 32 + lane) * 2;
                FragU bh, bl;
                bh.u[0] = sm.w1hi4[fb4];  bh.u[1] = sm.w1hi4[fb4 + 1];
                bl.u[0] = sm.w1lo4[fb4];  bl.u[1] = sm.w1lo4[fb4 + 1];
                v8f c = cacc[t];
                c = __builtin_amdgcn_wmma_f32_16x16x32_bf16(false, ah.v, false, bh.v, (short)0, c, false, false);
                c = __builtin_amdgcn_wmma_f32_16x16x32_bf16(false, ah.v, false, bl.v, (short)0, c, false, false);
                c = __builtin_amdgcn_wmma_f32_16x16x32_bf16(false, al.v, false, bh.v, (short)0, c, false, false);
                cacc[t] = c;
            }
        }

        // ---- sigmoid + row sums -> scores ----
        const int mrow0 = (lane >> 4) << 3;        // C layout: rows 0-7 / 8-15
        #pragma unroll
        for (int t = 0; t < 2; ++t) {
            const float bv = sm.bias[(wave * 2 + t) * 16 + (lane & 15)];
            #pragma unroll
            for (int i = 0; i < 8; ++i) {
                float x  = cacc[t][i] + bv;
                float sg = 1.0f / (1.0f + __expf(-x));
                #pragma unroll
                for (int off = 1; off < 16; off <<= 1)
                    sg += __shfl_xor(sg, off, 32);
                if ((lane & 15) == 0) atomicAdd(&sm.scores[mrow0 + i], sg);
            }
        }
        __syncthreads();

        // ---- mask + padding ----
        if (tid < 16) {
            int s = s0 + tid;
            float x = sm.scores[tid];
            if (s >= SS)                      x = -3.0e38f;   // pad row -> weight 0
            else if (mask[(long)b * SS + s])  x = -1.0e9f;    // matches reference
            sm.scores[tid] = x;
        }
        __syncthreads();

        // ---- online softmax + pooling from bf16 tile (seq = hi + lo) ----
        float tmax = -3.0e38f;
        #pragma unroll
        for (int s = 0; s < 16; ++s) tmax = fmaxf(tmax, sm.scores[s]);
        float mnew  = fmaxf(m, tmax);
        float scale = __expf(m - mnew);
        if (tid < 16) sm.wbuf[tid] = __expf(sm.scores[tid] - mnew);
        __syncthreads();
        const __bf16* shp = (const __bf16*)sm.sqhi4;
        const __bf16* slp = (const __bf16*)sm.sqlo4;
        float lsum = 0.0f, asum = 0.0f;
        #pragma unroll
        for (int s = 0; s < 16; ++s) {
            float w = sm.wbuf[s];
            lsum += w;
            asum = fmaf(w, (float)shp[s * DD + tid] + (float)slp[s * DD + tid], asum);
        }
        l   = l * scale + lsum;
        acc = acc * scale + asum;
        m   = mnew;

        if (st == STILES - 1) out[(long)b * DD + tid] = acc / l;
        __syncthreads();   // tile LDS + bias reused next iteration
    }
}

// ---------------------------------------------------------------------------
extern "C" void kernel_launch(void* const* d_in, const int* in_sizes, int n_in,
                              void* d_out, int out_size, void* d_ws, size_t ws_size,
                              hipStream_t stream) {
    const float*         cate = (const float*)d_in[0];
    const float*         seq  = (const float*)d_in[1];
    const unsigned char* mask = (const unsigned char*)d_in[2];
    const float*         W1   = (const float*)d_in[3];
    const float*         b1   = (const float*)d_in[4];
    const float*         W2   = (const float*)d_in[5];
    const float*         b2   = (const float*)d_in[6];
    float*               out  = (float*)d_out;

    // workspace: catep (4 MB) | W2T (256 KB) | W1 hi (128 KB) | W1 lo (128 KB, contiguous)
    float*  ws_catep = (float*)d_ws;
    float*  ws_w2t   = ws_catep + (long)BB * DD;
    __bf16* ws_w1hi  = (__bf16*)(ws_w2t + NW1);
    __bf16* ws_w1lo  = ws_w1hi + NW1;

    pack_w1_kernel<<<NW1 / 256, 256, 0, stream>>>(W1, ws_w1hi, ws_w1lo);
    transpose_w2_kernel<<<DD, DD, 0, stream>>>(W2, ws_w2t);
    catep_kernel<<<BB / 16, 256, 0, stream>>>(cate, b2, ws_w2t, ws_catep);
    fused_attn_kernel<<<BB / BPB, 256, 0, stream>>>(seq, mask, b1, ws_catep,
                                                    ws_w1hi, (const float*)ws_w1lo, out);
}
